// TopSim_52140902973493
// MI455X (gfx1250) — matrix operand order repaired and checked by reading, not compiled, not run on hardware
//
#include <hip/hip_runtime.h>

typedef __attribute__((ext_vector_type(16))) _Float16 v16h;
typedef __attribute__((ext_vector_type(8)))  _Float16 v8h;
typedef __attribute__((ext_vector_type(8)))  float    v8f;

#define N1   4096
#define N2   65536
#define D    64
#define TOPK 10

__device__ __forceinline__ v16h cat8(v8h a, v8h b) {
  return __builtin_shufflevector(a, b, 0,1,2,3,4,5,6,7,8,9,10,11,12,13,14,15);
}

// ---------------------------------------------------------------------------
// Kernel 1: fp32 -> normalized f16 rows (one thread per 64-elem row).
// ||x||^2 clamped >= 1e-8 so the norm product is clamped >= eps = 1e-8.
// ---------------------------------------------------------------------------
__global__ __launch_bounds__(256) void normf16_kernel(const float* __restrict__ x1,
                                                      const float* __restrict__ x2,
                                                      _Float16* __restrict__ x1h,
                                                      _Float16* __restrict__ x2h) {
  int r = blockIdx.x * blockDim.x + threadIdx.x;   // grid sized exactly N1+N2
  const float* src;
  _Float16*    dst;
  if (r < N1) { src = x1 + (size_t)r * D;        dst = x1h + (size_t)r * D; }
  else        { size_t rr = (size_t)(r - N1);
                src = x2 + rr * D;               dst = x2h + rr * D; }

  float v[D];
  float ss = 0.0f;
  #pragma unroll
  for (int i = 0; i < D / 4; ++i) {
    float4 f = ((const float4*)src)[i];
    v[4*i+0] = f.x; v[4*i+1] = f.y; v[4*i+2] = f.z; v[4*i+3] = f.w;
    ss += f.x*f.x + f.y*f.y + f.z*f.z + f.w*f.w;
  }
  float rn = rsqrtf(fmaxf(ss, 1e-8f));
  #pragma unroll
  for (int i = 0; i < D / 8; ++i) {
    v8h h;
    #pragma unroll
    for (int j = 0; j < 8; ++j) h[j] = (_Float16)(v[8*i + j] * rn);
    ((v8h*)dst)[i] = h;
  }
}

// ---------------------------------------------------------------------------
// Kernel 2: fused WMMA GEMM + per-query top-10 with packed u64 keys.
//   A (M side) = x2 tiles (streamed, double-buffered), B (N side) = 16 queries.
//   C layout => each lane owns one query: per-lane register top-10 of u64 keys
//   key = ordered(value):32 | ~index:32  (bigger key = better; ties -> low idx)
// ---------------------------------------------------------------------------
__global__ __launch_bounds__(256) void topsim_kernel(const _Float16* __restrict__ x1h,
                                                     const _Float16* __restrict__ x2h,
                                                     float* __restrict__ out_sim,
                                                     int* __restrict__ out_idx) {
  __shared__ unsigned long long sk[256 * TOPK];     // 20 KB of 320 KB LDS

  const int lane  = threadIdx.x & 31;
  const int wave  = threadIdx.x >> 5;
  const int col   = lane & 15;     // C column == query slot
  const int hi    = lane >> 4;     // upper/lower lane half
  const int qbase = blockIdx.x * 16;

  // B fragments (queries), constant across the entire N2 sweep.
  const _Float16* qrow = x1h + (size_t)(qbase + col) * D;
  v16h b0 = *(const v16h*)(qrow + hi * 16);        // K =  0..31
  v16h b1 = *(const v16h*)(qrow + 32 + hi * 16);   // K = 32..63

  // Init key encodes value = -inf (ordered(0xFF800000) = 0x007FFFFF).
  const unsigned long long KINIT = 0x007FFFFFull << 32;
  unsigned long long top[TOPK];
  #pragma unroll
  for (int j = 0; j < TOPK; ++j) top[j] = KINIT;
  float thr = -__builtin_inff();                   // value of current 10th best

  const int TILES = N2 / (16 * 8);                 // 512 tiles per wave

  // Prime the A-fragment double buffer (tile i = rows (wave + 8*i)*16 ...).
  const _Float16* p0 = x2h + (size_t)(wave * 16 + col) * D;
  v8h L0 = *(const v8h*)(p0 +       hi * 16);
  v8h L1 = *(const v8h*)(p0 + 8 +   hi * 16);
  v8h L2 = *(const v8h*)(p0 + 32 +  hi * 16);
  v8h L3 = *(const v8h*)(p0 + 40 +  hi * 16);

  for (int i = 0; i < TILES; ++i) {
    // Prefetch next tile's fragments before consuming the current ones.
    const int inext = (i + 1 == TILES) ? 0 : (i + 1);
    const _Float16* pn = x2h + (size_t)((wave + 8 * inext) * 16 + col) * D;
    v8h n0 = *(const v8h*)(pn +       hi * 16);
    v8h n1 = *(const v8h*)(pn + 8 +   hi * 16);
    v8h n2 = *(const v8h*)(pn + 32 +  hi * 16);
    v8h n3 = *(const v8h*)(pn + 40 +  hi * 16);

    v8f c = {};
    c = __builtin_amdgcn_wmma_f32_16x16x32_f16(false, cat8(L0, L1), false, b0,
                                               (short)0, c, false, false);
    c = __builtin_amdgcn_wmma_f32_16x16x32_f16(false, cat8(L2, L3), false, b1,
                                               (short)0, c, false, false);

    // Cheap tile-level filter: one branch per tile, not eight.
    float m = fmaxf(fmaxf(fmaxf(c[0], c[1]), fmaxf(c[2], c[3])),
                    fmaxf(fmaxf(c[4], c[5]), fmaxf(c[6], c[7])));
    if (m >= thr) {
      const int rb = (wave + 8 * i) * 16 + 8 * hi;   // this lane's index base
      #pragma unroll
      for (int r = 0; r < 8; ++r) {
        float v = c[r];
        if (v >= thr) {
          unsigned int bits = __float_as_uint(v);
          unsigned int ob   = bits ^ ((unsigned int)((int)bits >> 31) | 0x80000000u);
          unsigned long long k =
              ((unsigned long long)ob << 32) | (unsigned int)~(unsigned int)(rb + r);
          #pragma unroll
          for (int j = 0; j < TOPK; ++j) {           // branch-free push-down
            bool gt = k > top[j];
            unsigned long long t = top[j];
            top[j] = gt ? k : t;
            k      = gt ? t : k;
          }
          unsigned int ov = (unsigned int)(top[TOPK - 1] >> 32);
          unsigned int vb = (ov & 0x80000000u) ? (ov ^ 0x80000000u) : ~ov;
          thr = __uint_as_float(vb);
        }
      }
    }
    L0 = n0; L1 = n1; L2 = n2; L3 = n3;
  }

  // Dump per-lane sorted key lists to LDS.
  #pragma unroll
  for (int j = 0; j < TOPK; ++j) sk[threadIdx.x * TOPK + j] = top[j];
  __syncthreads();

  // Threads 0..15: query q = tid owns lists of threads with (tid & 15) == q.
  // Keys are unique (disjoint index partitions), so a strict threshold scan
  // yields the global top-10 with no pointer state.
  if (threadIdx.x < 16) {
    const int q = threadIdx.x;
    unsigned long long prev = ~0ull;
    for (int out = 0; out < TOPK; ++out) {
      unsigned long long best = 0;
      for (int l = 0; l < 16; ++l) {
        const int srcb = (q + l * 16) * TOPK;
        #pragma unroll
        for (int j = 0; j < TOPK; ++j) {
          unsigned long long k = sk[srcb + j];
          if (k < prev && k > best) best = k;
        }
      }
      unsigned int ov = (unsigned int)(best >> 32);
      unsigned int vb = (ov & 0x80000000u) ? (ov ^ 0x80000000u) : ~ov;
      out_sim[(qbase + q) * TOPK + out] = __uint_as_float(vb);
      out_idx[(qbase + q) * TOPK + out] = (int)~(unsigned int)best;
      prev = best;
    }
  }
}

// ---------------------------------------------------------------------------
extern "C" void kernel_launch(void* const* d_in, const int* in_sizes, int n_in,
                              void* d_out, int out_size, void* d_ws, size_t ws_size,
                              hipStream_t stream) {
  const float* x1 = (const float*)d_in[0];   // [4096, 64] f32
  const float* x2 = (const float*)d_in[1];   // [65536, 64] f32
  // d_in[2] = n (== 10), fixed at compile time.

  _Float16* x1h = (_Float16*)d_ws;                   // 512 KB
  _Float16* x2h = x1h + (size_t)N1 * D;              // 8 MB
  float* out_sim = (float*)d_out;                    // [4096,10] f32
  int*   out_idx = (int*)d_out + (size_t)N1 * TOPK;  // [4096,10] i32

  normf16_kernel<<<(N1 + N2) / 256, 256, 0, stream>>>(x1, x2, x1h, x2h);
  topsim_kernel<<<N1 / 16, 256, 0, stream>>>(x1h, x2h, out_sim, out_idx);
}
